// NeighborhoodAttentionGeneric_32203664785483
// MI455X (gfx1250) — matrix operand order repaired and checked by reading, not compiled, not run on hardware
//
#include <hip/hip_runtime.h>
#include <hip/hip_bf16.h>

typedef __attribute__((ext_vector_type(16))) _Float16 v16h;
typedef __attribute__((ext_vector_type(8)))  _Float16 h8;
typedef __attribute__((ext_vector_type(8)))  float    v8f;
typedef __attribute__((ext_vector_type(4)))  int      v4i;

#define BDIM 2
#define HDIM 48
#define WDIM 48
#define CDIM 512
#define NHEAD 8
#define HD 64
#define KK 7
#define MTOT (BDIM*HDIM*WDIM)      // 4608
#define NQKV (3*CDIM)              // 1536
#define KDEPTH CDIM                // 512

// LDS staging geometry (per stage): A tile 64x64 f16, B tile 64x256 f16
#define KSTAGE   64
#define A_STRIDE 72                 // 64 + 8 halfs pad (144B rows, 16B aligned)
#define B_STRIDE 264                // 256 + 8 halfs pad (528B rows, 16B aligned)
#define A_ELEMS  (64 * A_STRIDE)
#define B_ELEMS  (64 * B_STRIDE)

// ---------------- async global->LDS (CDNA5 ASYNCcnt path) ---------------
typedef __attribute__((address_space(1))) v4i* gv4i_p;   // global 128-bit chunk
typedef __attribute__((address_space(3))) v4i* sv4i_p;   // LDS 128-bit chunk

#if __has_builtin(__builtin_amdgcn_global_load_async_to_lds_b128)
#define HAS_ASYNC_LDS 1
__device__ __forceinline__ void async_cp16(const _Float16* g, _Float16* l) {
    __builtin_amdgcn_global_load_async_to_lds_b128((gv4i_p)g, (sv4i_p)l, 0, 0);
}
__device__ __forceinline__ void async_wait_all() {
#if __has_builtin(__builtin_amdgcn_s_wait_asynccnt)
    __builtin_amdgcn_s_wait_asynccnt(0);
#else
    asm volatile("s_wait_asynccnt 0x0" ::: "memory");
#endif
}
#else
#define HAS_ASYNC_LDS 0
__device__ __forceinline__ void async_cp16(const _Float16* g, _Float16* l) {
    *(h8*)l = *(const h8*)g;           // sync fallback
}
__device__ __forceinline__ void async_wait_all() {}
#endif

__device__ __forceinline__ v8f wmma_f16(v16h a, v16h b, v8f c) {
    return __builtin_amdgcn_wmma_f32_16x16x32_f16(false, a, false, b, (short)0, c,
                                                  false, false);
}

// ---------------- stage issue: 256 threads cooperatively copy tiles -----
// A: (MxK row-major) 64 rows x 64 halfs = 512 16B-chunks -> 2/thread
// B: (KxN K-major)   64 rows x 256 halfs = 2048 chunks   -> 8/thread
__device__ __forceinline__ void stage_issue(const _Float16* __restrict__ A,
                                            const _Float16* __restrict__ BT, int ldb,
                                            int mblk, int nblk, int k0,
                                            _Float16* dA, _Float16* dB, int tid) {
    #pragma unroll
    for (int i = 0; i < 2; ++i) {
        int c   = tid + 256 * i;
        int row = c >> 3, ch = (c & 7) << 3;
        async_cp16(A + (size_t)(mblk + row) * KDEPTH + k0 + ch,
                   dA + row * A_STRIDE + ch);
    }
    #pragma unroll
    for (int i = 0; i < 8; ++i) {
        int c   = tid + 256 * i;
        int row = c >> 5, nch = (c & 31) << 3;
        async_cp16(BT + (size_t)(k0 + row) * ldb + nblk + nch,
                   dB + row * B_STRIDE + nch);
    }
}

// ---------------- consume one stage: 2 K-chunks x 8 WMMAs ---------------
__device__ __forceinline__ void consume_stage(const _Float16* sA, const _Float16* sB,
                                              int wm, int wn, int lane,
                                              v8f acc[2][4]) {
    #pragma unroll
    for (int kc = 0; kc < KSTAGE; kc += 32) {
        v16h a[2];
        #pragma unroll
        for (int im = 0; im < 2; ++im) {
            int ml = wm * 32 + im * 16 + (lane & 15);
            int hi = (lane >> 4) << 3;
            const _Float16* p = sA + ml * A_STRIDE + kc + hi;
            union { v16h v; h8 h[2]; } f;
            f.h[0] = *(const h8*)p;
            f.h[1] = *(const h8*)(p + 16);
            a[im] = f.v;
        }
        #pragma unroll
        for (int t = 0; t < 4; ++t) {
            const _Float16* p = sB + (kc + ((lane >> 4) << 3)) * B_STRIDE + wn * 64 + 16 * t + (lane & 15);
            v16h b;
            #pragma unroll
            for (int s = 0; s < 8; ++s) {
                b[s]     = p[s * B_STRIDE];
                b[8 + s] = p[(16 + s) * B_STRIDE];
            }
            acc[0][t] = wmma_f16(a[0], b, acc[0][t]);
            acc[1][t] = wmma_f16(a[1], b, acc[1][t]);
        }
    }
}

// ---------------- double-buffered GEMM core ------------------------------
__device__ __forceinline__ void gemm_core(const _Float16* __restrict__ A,
                                          const _Float16* __restrict__ BT, int ldb,
                                          int mblk, int nblk,
                                          _Float16* sA, _Float16* sB,   // [2][*]
                                          v8f acc[2][4], int lane, int wave, int tid) {
    const int wm = wave & 1, wn = wave >> 1;
    const int NIT = KDEPTH / KSTAGE;   // 8
    stage_issue(A, BT, ldb, mblk, nblk, 0, sA, sB, tid);
    async_wait_all();
    __syncthreads();
    for (int it = 0; it < NIT; ++it) {
        int cur = it & 1;
        if (it + 1 < NIT)
            stage_issue(A, BT, ldb, mblk, nblk, (it + 1) * KSTAGE,
                        sA + (cur ^ 1) * A_ELEMS, sB + (cur ^ 1) * B_ELEMS, tid);
        consume_stage(sA + cur * A_ELEMS, sB + cur * B_ELEMS, wm, wn, lane, acc);
        async_wait_all();
        __syncthreads();
    }
}

// ---------------- converters -------------------------------------------
__global__ __launch_bounds__(256) void cvt_f16(const float* __restrict__ s,
                                               _Float16* __restrict__ d, int n) {
    int id = blockIdx.x * 256 + threadIdx.x;
    if (id < n) d[id] = (_Float16)s[id];
}

// (rows x cols) f32 row-major  ->  (cols x rows) f16 (K-major weights)
__global__ __launch_bounds__(256) void transpose_cvt(const float* __restrict__ s,
                                                     _Float16* __restrict__ d,
                                                     int rows, int cols) {
    int id = blockIdx.x * 256 + threadIdx.x;
    int r = id / cols, c = id % cols;
    d[(size_t)c * rows + r] = (_Float16)s[id];
}

// ---------------- GEMM 1: qkv = x @ Wqkv^T + b, split & scale ----------
__global__ __launch_bounds__(256) void gemm_qkv(const _Float16* __restrict__ xh,
                                                const _Float16* __restrict__ wT,
                                                const float* __restrict__ bias,
                                                _Float16* __restrict__ qb,
                                                _Float16* __restrict__ kb,
                                                _Float16* __restrict__ vb) {
    __shared__ _Float16 sA[2 * A_ELEMS];
    __shared__ _Float16 sB[2 * B_ELEMS];
    const int tid = threadIdx.x, lane = tid & 31, wave = tid >> 5;
    const int mblk = blockIdx.x * 64;
    const int nblk = blockIdx.y * 256;

    const v8f vzero = {0.f,0.f,0.f,0.f,0.f,0.f,0.f,0.f};
    v8f acc[2][4];
    #pragma unroll
    for (int im = 0; im < 2; ++im)
        #pragma unroll
        for (int t = 0; t < 4; ++t) acc[im][t] = vzero;

    gemm_core(xh, wT, NQKV, mblk, nblk, sA, sB, acc, lane, wave, tid);

    // epilogue: C/D layout -> M = r + 8*(lane>=16), N = lane&15
    const int wm = wave & 1, wn = wave >> 1;
    const int m0 = mblk + wm * 32, n0 = nblk + wn * 64;
    const int rbase = (lane >> 4) << 3;
    const int nl = lane & 15;
    #pragma unroll
    for (int im = 0; im < 2; ++im)
        #pragma unroll
        for (int t = 0; t < 4; ++t)
            #pragma unroll
            for (int r = 0; r < 8; ++r) {
                int m = m0 + im * 16 + rbase + r;
                int n = n0 + 16 * t + nl;
                float val = acc[im][t][r] + bias[n];
                int slot = n >> 9;          // 0=q 1=k 2=v
                int c    = n & 511;
                _Float16* dst = (slot == 0) ? qb : (slot == 1 ? kb : vb);
                if (slot == 0) val *= 0.125f;       // hd^-0.5, hd=64
                dst[(size_t)m * CDIM + c] = (_Float16)val;
            }
}

// ---------------- attention: one wave32 per (query, head) ---------------
__global__ __launch_bounds__(256) void attn_kernel(const _Float16* __restrict__ q,
                                                   const _Float16* __restrict__ kkey,
                                                   const _Float16* __restrict__ v,
                                                   _Float16* __restrict__ o) {
    const int lane = threadIdx.x & 31;
    const int wave = threadIdx.x >> 5;
    const int gid  = blockIdx.x * 8 + wave;        // q_idx*8 + head
    const int h    = gid & 7;
    const int qidx = gid >> 3;                     // (b*48+i)*48 + j
    const int j  = qidx % WDIM;
    const int bi = qidx / WDIM;
    const int i  = bi % HDIM;
    const int ri0 = min(max(i - 3, 0), HDIM - KK);
    const int cj0 = min(max(j - 3, 0), WDIM - KK);
    const int browbase = (bi / HDIM) * HDIM;       // b*48

    const _Float16* qp = q + (size_t)qidx * CDIM + h * HD;

    // phase 1: logits, lanes own window positions p = lane, lane+32
    float l0 = -1e30f, l1 = -1e30f;
    #pragma unroll
    for (int t = 0; t < 2; ++t) {
        int p = lane + 32 * t;
        if (p < KK * KK) {
            int pr = p / KK, pc = p % KK;
            int kidx = (browbase + ri0 + pr) * WDIM + (cj0 + pc);
            const _Float16* kp = kkey + (size_t)kidx * CDIM + h * HD;
            const h8* qv = (const h8*)qp;
            const h8* kv = (const h8*)kp;
            float acc = 0.f;
            #pragma unroll
            for (int c8 = 0; c8 < 8; ++c8) {
                h8 qq = qv[c8], kq = kv[c8];
                #pragma unroll
                for (int e = 0; e < 8; ++e) acc += (float)qq[e] * (float)kq[e];
            }
            if (t == 0) l0 = acc; else l1 = acc;
        }
    }
    // softmax across the wave
    float m = fmaxf(l0, l1);
    #pragma unroll
    for (int off = 16; off > 0; off >>= 1) m = fmaxf(m, __shfl_xor(m, off, 32));
    float e0 = (l0 > -1e29f) ? __expf(l0 - m) : 0.f;
    float e1 = (l1 > -1e29f) ? __expf(l1 - m) : 0.f;
    float s = e0 + e1;
    #pragma unroll
    for (int off = 16; off > 0; off >>= 1) s += __shfl_xor(s, off, 32);
    float inv = 1.f / s;
    float w0 = e0 * inv, w1 = e1 * inv;

    // phase 2: lanes own head-dims d = lane, lane+32
    float o0 = 0.f, o1 = 0.f;
    for (int p = 0; p < KK * KK; ++p) {
        float wp = (p < 32) ? __shfl(w0, p, 32) : __shfl(w1, p - 32, 32);
        int pr = p / KK, pc = p % KK;
        int kidx = (browbase + ri0 + pr) * WDIM + (cj0 + pc);
        const _Float16* vp = v + (size_t)kidx * CDIM + h * HD;
        o0 += wp * (float)vp[lane];
        o1 += wp * (float)vp[lane + 32];
    }
    _Float16* op = o + (size_t)qidx * CDIM + h * HD;
    op[lane]      = (_Float16)o0;
    op[lane + 32] = (_Float16)o1;
}

// ---------------- GEMM 2: out = ao @ Wproj^T + b (f32 out) --------------
__global__ __launch_bounds__(256) void gemm_proj(const _Float16* __restrict__ ah,
                                                 const _Float16* __restrict__ wT,
                                                 const float* __restrict__ bias,
                                                 float* __restrict__ out) {
    __shared__ _Float16 sA[2 * A_ELEMS];
    __shared__ _Float16 sB[2 * B_ELEMS];
    const int tid = threadIdx.x, lane = tid & 31, wave = tid >> 5;
    const int mblk = blockIdx.x * 64;
    const int nblk = blockIdx.y * 256;

    const v8f vzero = {0.f,0.f,0.f,0.f,0.f,0.f,0.f,0.f};
    v8f acc[2][4];
    #pragma unroll
    for (int im = 0; im < 2; ++im)
        #pragma unroll
        for (int t = 0; t < 4; ++t) acc[im][t] = vzero;

    gemm_core(ah, wT, CDIM, mblk, nblk, sA, sB, acc, lane, wave, tid);

    const int wm = wave & 1, wn = wave >> 1;
    const int m0 = mblk + wm * 32, n0 = nblk + wn * 64;
    const int rbase = (lane >> 4) << 3;
    const int nl = lane & 15;
    #pragma unroll
    for (int im = 0; im < 2; ++im)
        #pragma unroll
        for (int t = 0; t < 4; ++t)
            #pragma unroll
            for (int r = 0; r < 8; ++r) {
                int m = m0 + im * 16 + rbase + r;
                int n = n0 + 16 * t + nl;
                out[(size_t)m * CDIM + n] = acc[im][t][r] + bias[n];
            }
}

// ------------------------------------------------------------------------
extern "C" void kernel_launch(void* const* d_in, const int* in_sizes, int n_in,
                              void* d_out, int out_size, void* d_ws, size_t ws_size,
                              hipStream_t stream) {
    const float* x      = (const float*)d_in[0];
    const float* qkv_w  = (const float*)d_in[1];
    const float* qkv_b  = (const float*)d_in[2];
    const float* proj_w = (const float*)d_in[3];
    const float* proj_b = (const float*)d_in[4];
    float* out = (float*)d_out;

    char* ws = (char*)d_ws;
    size_t off = 0;
    _Float16* xh     = (_Float16*)(ws + off); off += (size_t)MTOT * CDIM * 2;
    _Float16* wqkvT  = (_Float16*)(ws + off); off += (size_t)KDEPTH * NQKV * 2;
    _Float16* wprojT = (_Float16*)(ws + off); off += (size_t)KDEPTH * CDIM * 2;
    _Float16* qb     = (_Float16*)(ws + off); off += (size_t)MTOT * CDIM * 2;
    _Float16* kb     = (_Float16*)(ws + off); off += (size_t)MTOT * CDIM * 2;
    _Float16* vb     = (_Float16*)(ws + off); off += (size_t)MTOT * CDIM * 2;
    _Float16* ao     = (_Float16*)(ws + off); off += (size_t)MTOT * CDIM * 2;

    // converts
    cvt_f16<<<(MTOT * CDIM) / 256, 256, 0, stream>>>(x, xh, MTOT * CDIM);
    transpose_cvt<<<(NQKV * KDEPTH) / 256, 256, 0, stream>>>(qkv_w, wqkvT, NQKV, KDEPTH);
    transpose_cvt<<<(CDIM * KDEPTH) / 256, 256, 0, stream>>>(proj_w, wprojT, CDIM, KDEPTH);

    // qkv GEMM (WMMA + async-LDS double buffering)
    dim3 g1(MTOT / 64, NQKV / 256);
    gemm_qkv<<<g1, 256, 0, stream>>>(xh, wqkvT, qkv_b, qb, kb, vb);

    // neighborhood attention (wave per query-head)
    attn_kernel<<<(MTOT * NHEAD) / 8, 256, 0, stream>>>(qb, kb, vb, ao);

    // output projection GEMM (WMMA + async-LDS double buffering)
    dim3 g2(MTOT / 64, CDIM / 256);
    gemm_proj<<<g2, 256, 0, stream>>>(ao, wprojT, proj_b, out);
}
